// IGMTF_3049426780655
// MI455X (gfx1250) — compile-verified, matrix operand order checked by reading, not run on hardware
//
#include <hip/hip_runtime.h>
#include <hip/hip_bf16.h>
#include <math.h>

// ---------- constants ----------
#define D_FEAT 6
#define HID 256
#define T_STEPS 60
#define B_DAYS 4
#define STOCKS 512
#define N_SEQ (B_DAYS * STOCKS)        // 2048
#define G3 (3 * HID)                   // 768
#define TRAIN_DAYS 240
#define K_DAY 10
#define N_NEIGHBOR 10
#define N_SAMPLE (B_DAYS * K_DAY * STOCKS)  // 20480
#define SLOPE 0.01f

typedef __attribute__((ext_vector_type(16))) _Float16 v16h;
typedef __attribute__((ext_vector_type(8)))  _Float16 v8h;
typedef __attribute__((ext_vector_type(8)))  float    v8f;

// ---------- WMMA helpers ----------
// 16-bit A (16x32) fragment layout (ISA 7.12.2): lane L<16 holds row L,
// K = {0..7, 16..23}; lane L+16 holds row L, K = {8..15, 24..31}.
// B (32x16) loaded from a row-major (N x K) matrix uses the identical pattern
// (row = output column). Load as two contiguous 16B chunks.
__device__ __forceinline__ v16h ldfrag(const _Float16* p, int lane) {
  const _Float16* q = p + ((lane & 16) ? 8 : 0);
  v8h lo = *(const v8h*)(q);
  v8h hi = *(const v8h*)(q + 16);
  return __builtin_shufflevector(lo, hi, 0,1,2,3,4,5,6,7,8,9,10,11,12,13,14,15);
}

__device__ __forceinline__ v8f wmma_acc(v16h a, v16h b, v8f c) {
  return __builtin_amdgcn_wmma_f32_16x16x32_f16(false, a, false, b, (short)0, c,
                                                false, false);
}

__device__ __forceinline__ float sigm(float x) { return 1.f / (1.f + __expf(-x)); }

// ---------- f32 -> f16 convert ----------
__global__ void k_f32_to_f16(const float* __restrict__ s, _Float16* __restrict__ d, int n) {
  int i = blockIdx.x * 256 + threadIdx.x;
  if (i < n) d[i] = (_Float16)s[i];
}

// ---------- GRU layer 0 ----------
// grid.x = N_SEQ/16, block = 256 (8 waves). dyn LDS: h16(8K) h32(16K) gh(48K)
__global__ void k_gru0(const float* __restrict__ inp,       // (N, 360)
                       const _Float16* __restrict__ whh16,  // (768, 256) f16
                       const float* __restrict__ wih,       // (768, 6)
                       const float* __restrict__ b_ih,      // (768)
                       const float* __restrict__ b_hh,      // (768)
                       _Float16* __restrict__ hseq)         // (N, T, 256) f16
{
  extern __shared__ char smem[];
  _Float16* h16 = (_Float16*)smem;                    // 16*256 f16
  float*    h32 = (float*)(smem + 8192);              // 16*256 f32
  float*    gh  = (float*)(smem + 8192 + 16384);      // 16*768 f32

  const int n0 = blockIdx.x * 16;
  const int tid = threadIdx.x;
  const int wave = tid >> 5, lane = tid & 31, row = lane & 15;

  for (int e = tid; e < 16 * HID; e += 256) { h32[e] = 0.f; h16[e] = (_Float16)0.f; }
  __syncthreads();

  for (int t = 0; t < T_STEPS; ++t) {
    // gh = h(16x256) @ W_hh.T -> (16x768) via WMMA
    for (int gt = wave; gt < 48; gt += 8) {
      v8f acc = {};
      for (int k0 = 0; k0 < HID; k0 += 32) {
        v16h a = ldfrag(h16 + row * HID + k0, lane);
        v16h b = ldfrag(whh16 + (size_t)(gt * 16 + row) * HID + k0, lane);
        acc = wmma_acc(a, b, acc);
      }
      int mb = (lane & 16) ? 8 : 0, n = lane & 15;
#pragma unroll
      for (int v = 0; v < 8; ++v) gh[(mb + v) * G3 + gt * 16 + n] = acc[v];
    }
    __syncthreads();

    // gates (layer-0 input projection is K=6: do it inline on VALU)
    for (int e = tid; e < 16 * HID; e += 256) {
      int m = e >> 8, c = e & 255;
      const float* xrow = inp + (size_t)(n0 + m) * (D_FEAT * T_STEPS);
      float xr = b_ih[c], xz = b_ih[HID + c], xn = b_ih[2 * HID + c];
#pragma unroll
      for (int i = 0; i < D_FEAT; ++i) {
        float xv = xrow[i * T_STEPS + t];
        xr += xv * wih[c * D_FEAT + i];
        xz += xv * wih[(HID + c) * D_FEAT + i];
        xn += xv * wih[(2 * HID + c) * D_FEAT + i];
      }
      float hr = gh[m * G3 + c] + b_hh[c];
      float hz = gh[m * G3 + HID + c] + b_hh[HID + c];
      float hn = gh[m * G3 + 2 * HID + c] + b_hh[2 * HID + c];
      float r = sigm(xr + hr), z = sigm(xz + hz);
      float nn = tanhf(xn + r * hn);
      float hnew = (1.f - z) * nn + z * h32[e];
      h32[e] = hnew;
      h16[e] = (_Float16)hnew;
      hseq[((size_t)(n0 + m) * T_STEPS + t) * HID + c] = (_Float16)hnew;
    }
    __syncthreads();
  }
}

// ---------- GRU layer 1 ----------
// Per timestep also computes the input projection x@W_ih1.T via WMMA (xw never
// materialized in global memory). dyn LDS: h16(8K) h32(16K) gh(48K) xw(48K)
__global__ void k_gru1(const _Float16* __restrict__ hseq,   // (N, T, 256) f16
                       const _Float16* __restrict__ wih16,  // (768, 256) f16
                       const _Float16* __restrict__ whh16,  // (768, 256) f16
                       const float* __restrict__ b_ih,
                       const float* __restrict__ b_hh,
                       _Float16* __restrict__ out16)        // (N, 256) f16
{
  extern __shared__ char smem[];
  _Float16* h16 = (_Float16*)smem;
  float*    h32 = (float*)(smem + 8192);
  float*    gh  = (float*)(smem + 8192 + 16384);
  float*    xw  = (float*)(smem + 8192 + 16384 + 49152);

  const int n0 = blockIdx.x * 16;
  const int tid = threadIdx.x;
  const int wave = tid >> 5, lane = tid & 31, row = lane & 15;

  for (int e = tid; e < 16 * HID; e += 256) { h32[e] = 0.f; h16[e] = (_Float16)0.f; }
  __syncthreads();

  for (int t = 0; t < T_STEPS; ++t) {
    // prefetch next timestep's input rows into cache while we compute
    if (t + 1 < T_STEPS)
      __builtin_prefetch(hseq + ((size_t)(n0 + row) * T_STEPS + t + 1) * HID, 0, 2);

    // 96 tile jobs: j<48 -> xw tile (A = hseq rows at time t), else gh tile
    for (int j = wave; j < 96; j += 8) {
      int gt = j % 48;
      bool isxw = (j < 48);
      v8f acc = {};
      const _Float16* brow = (isxw ? wih16 : whh16) + (size_t)(gt * 16 + row) * HID;
      const _Float16* arow = isxw
          ? hseq + ((size_t)(n0 + row) * T_STEPS + t) * HID
          : h16 + row * HID;
      for (int k0 = 0; k0 < HID; k0 += 32) {
        v16h a = ldfrag(arow + k0, lane);
        v16h b = ldfrag(brow + k0, lane);
        acc = wmma_acc(a, b, acc);
      }
      float* dst = isxw ? xw : gh;
      int mb = (lane & 16) ? 8 : 0, n = lane & 15;
#pragma unroll
      for (int v = 0; v < 8; ++v) dst[(mb + v) * G3 + gt * 16 + n] = acc[v];
    }
    __syncthreads();

    for (int e = tid; e < 16 * HID; e += 256) {
      int m = e >> 8, c = e & 255;
      float xr = xw[m * G3 + c] + b_ih[c];
      float xz = xw[m * G3 + HID + c] + b_ih[HID + c];
      float xn = xw[m * G3 + 2 * HID + c] + b_ih[2 * HID + c];
      float hr = gh[m * G3 + c] + b_hh[c];
      float hz = gh[m * G3 + HID + c] + b_hh[HID + c];
      float hn = gh[m * G3 + 2 * HID + c] + b_hh[2 * HID + c];
      float r = sigm(xr + hr), z = sigm(xz + hz);
      float nn = tanhf(xn + r * hn);
      float hnew = (1.f - z) * nn + z * h32[e];
      h32[e] = hnew;
      h16[e] = (_Float16)hnew;
      if (t == T_STEPS - 1) out16[(size_t)(n0 + m) * HID + c] = (_Float16)hnew;
    }
    __syncthreads();
  }
}

// ---------- generic WMMA GEMM: C = act(A @ B.T + bias) ----------
// A: (M x K) f16 row-major; B: (G x K) f16 row-major; grid = (M/16, G/128),
// block = 256 (8 waves, one 16x16 tile per wave). ACT: 0=none, 1=leaky.
template <int ACT>
__global__ void k_gemm16(const _Float16* __restrict__ A,
                         const _Float16* __restrict__ B, int K,
                         const float* __restrict__ bias,
                         float* __restrict__ Cf, _Float16* __restrict__ Ch,
                         int ldc)
{
  const int m0 = blockIdx.x * 16;
  const int wave = threadIdx.x >> 5, lane = threadIdx.x & 31, row = lane & 15;
  const int g0 = (blockIdx.y * 8 + wave) * 16;

  v8f acc = {};
  const _Float16* arow = A + (size_t)(m0 + row) * K;
  const _Float16* brow = B + (size_t)(g0 + row) * K;
  for (int k0 = 0; k0 < K; k0 += 32) {
    v16h a = ldfrag(arow + k0, lane);
    v16h b = ldfrag(brow + k0, lane);
    acc = wmma_acc(a, b, acc);
  }
  const int mb = (lane & 16) ? 8 : 0, n = lane & 15;
  const float bv = bias ? bias[g0 + n] : 0.f;
#pragma unroll
  for (int v = 0; v < 8; ++v) {
    float x = acc[v] + bv;
    if (ACT == 1) x = (x >= 0.f) ? x : SLOPE * x;
    size_t o = (size_t)(m0 + mb + v) * ldc + g0 + n;
    if (Cf) Cf[o] = x;
    if (Ch) Ch[o] = (_Float16)x;
  }
}

// ---------- day mean + cosine sim + top-K_DAY ----------
__global__ void k_day_topk(const float* __restrict__ mb,   // (2048, 256) f32
                           const float* __restrict__ thd,  // (240, 256) f32
                           int* __restrict__ day_idx)      // (4, 10)
{
  __shared__ float md[HID];
  __shared__ float sim[TRAIN_DAYS];
  __shared__ float mbn;
  const int b = blockIdx.x, t = threadIdx.x;
  if (t < HID) {
    float s = 0.f;
    for (int st = 0; st < STOCKS; ++st)
      s += mb[(size_t)(b * STOCKS + st) * HID + t];
    md[t] = s * (1.f / STOCKS);
  }
  __syncthreads();
  if (t == 0) {
    float s = 0.f;
    for (int c = 0; c < HID; ++c) s += md[c] * md[c];
    mbn = sqrtf(s);
  }
  __syncthreads();
  if (t < TRAIN_DAYS) {
    float dot = 0.f, ns = 0.f;
    for (int c = 0; c < HID; ++c) {
      float v = thd[(size_t)t * HID + c];
      dot += md[c] * v;
      ns += v * v;
    }
    float den = mbn * sqrtf(ns);
    float s = (den > 0.f) ? dot / den : 0.f;
    if (!(s == s)) s = 0.f;
    sim[t] = s;
  }
  __syncthreads();
  if (t == 0) {
    for (int k = 0; k < K_DAY; ++k) {
      int bi = 0; float bvv = -1e30f;
      for (int d = 0; d < TRAIN_DAYS; ++d)
        if (sim[d] > bvv) { bvv = sim[d]; bi = d; }
      sim[bi] = -1e30f;
      day_idx[b * K_DAY + k] = bi;
    }
  }
}

// ---------- gather sampled train hidden, convert to f16 ----------
__global__ void k_gather(const float* __restrict__ th,     // (240, 512, 256)
                         const int* __restrict__ day_idx,  // (40)
                         _Float16* __restrict__ dst)       // (20480, 256) f16
{
  size_t i = (size_t)blockIdx.x * 256 + threadIdx.x;
  if (i >= (size_t)N_SAMPLE * HID) return;
  int c = (int)(i & 255);
  size_t r = i >> 8;
  int s = (int)(r & (STOCKS - 1));
  int bk = (int)(r >> 9);  // 0..39
  int d = day_idx[bk];
  dst[i] = (_Float16)th[((size_t)d * STOCKS + s) * HID + c];
}

// ---------- row L2 norms ----------
__global__ void k_rownorm(const float* __restrict__ X, float* __restrict__ nrm,
                          int K, int rows) {
  int r = blockIdx.x * blockDim.x + threadIdx.x;
  if (r < rows) {
    float s = 0.f;
    for (int c = 0; c < K; ++c) { float v = X[(size_t)r * K + c]; s += v * v; }
    nrm[r] = sqrtf(s);
  }
}

// ---------- streaming cosine top-k + aggregation + final FC ----------
// grid.x = 2048/16; block = 256 (8 waves). Never materializes the 2048x20480
// score matrix: 16x256 WMMA tiles -> LDS. Top-10 maintained 16-way parallel
// per row (thread t owns row t>>4, column slice (t&15)*16), merged at end.
__global__ void k_topk_agg(const _Float16* __restrict__ q16,  // (2048, 256)
                           const _Float16* __restrict__ kh16, // (20480, 256)
                           const float* __restrict__ khf,     // (20480, 256)
                           const float* __restrict__ qn,      // (2048)
                           const float* __restrict__ khn,     // (20480)
                           const float* __restrict__ mbf,     // (2048, 256)
                           const float* __restrict__ fcw,     // (512)
                           const float* __restrict__ fcb,     // (1)
                           float* __restrict__ out)           // (2048)
{
  __shared__ alignas(16) _Float16 qtile[16 * HID];  // 8 KB
  __shared__ float stile[16 * HID];                 // 16 KB (scores, later agg)
  __shared__ float candv[256 * N_NEIGHBOR];         // 10 KB
  __shared__ int   candi[256 * N_NEIGHBOR];         // 10 KB
  __shared__ float topv[16 * N_NEIGHBOR];
  __shared__ int   topi[16 * N_NEIGHBOR];

  const int m0 = blockIdx.x * 16;
  const int tid = threadIdx.x;
  const int wave = tid >> 5, lane = tid & 31, row = lane & 15;

  // stage q tile (f16) into LDS
  {
    int m = tid >> 4, c0 = (tid & 15) * 16;
    for (int j = 0; j < 16; ++j)
      qtile[m * HID + c0 + j] = q16[(size_t)(m0 + m) * HID + c0 + j];
  }
  __syncthreads();

  // per-thread partial top-10 over its column slice
  float tv[N_NEIGHBOR];
  int ti[N_NEIGHBOR];
#pragma unroll
  for (int k = 0; k < N_NEIGHBOR; ++k) { tv[k] = -1e30f; ti[k] = 0; }

  for (int chunk = 0; chunk < N_SAMPLE; chunk += 256) {
    // 16 column tiles per chunk; wave handles 2
    for (int tt = 0; tt < 2; ++tt) {
      int ct = wave * 2 + tt;
      int gc0 = chunk + ct * 16;
      if (chunk + 256 < N_SAMPLE)
        __builtin_prefetch(kh16 + (size_t)(gc0 + 256 + row) * HID, 0, 2);
      v8f acc = {};
      for (int k0 = 0; k0 < HID; k0 += 32) {
        v16h a = ldfrag(qtile + row * HID + k0, lane);
        v16h b = ldfrag(kh16 + (size_t)(gc0 + row) * HID + k0, lane);
        acc = wmma_acc(a, b, acc);
      }
      int mb = (lane & 16) ? 8 : 0, n = lane & 15;
      float dn = khn[gc0 + n];
#pragma unroll
      for (int v = 0; v < 8; ++v) {
        int m = mb + v;
        float den = qn[m0 + m] * dn;
        float s = (den > 0.f) ? acc[v] / den : 0.f;
        if (!(s == s)) s = 0.f;
        stile[m * HID + ct * 16 + n] = s;
      }
    }
    __syncthreads();
    // 16-way parallel scan per row: thread owns 16 columns of its row
    {
      int m = tid >> 4, cbase = (tid & 15) * 16;
      for (int j = 0; j < 16; ++j) {
        float s = stile[m * HID + cbase + j];
        if (s > tv[N_NEIGHBOR - 1]) {
          int p = N_NEIGHBOR - 1;
          while (p > 0 && tv[p - 1] < s) {
            tv[p] = tv[p - 1]; ti[p] = ti[p - 1]; --p;
          }
          tv[p] = s; ti[p] = chunk + cbase + j;
        }
      }
    }
    __syncthreads();
  }

  // dump per-thread candidates; merge 160 -> 10 per row
#pragma unroll
  for (int k = 0; k < N_NEIGHBOR; ++k) {
    candv[tid * N_NEIGHBOR + k] = tv[k];
    candi[tid * N_NEIGHBOR + k] = ti[k];
  }
  __syncthreads();
  if (tid < 16) {
    int m = tid;
    for (int k = 0; k < N_NEIGHBOR; ++k) {
      float best = -1e30f; int bp = 0;
      for (int c = 0; c < 16 * N_NEIGHBOR; ++c) {
        float v = candv[m * 16 * N_NEIGHBOR + c];
        if (v > best) { best = v; bp = c; }
      }
      topv[m * N_NEIGHBOR + k] = best;
      topi[m * N_NEIGHBOR + k] = candi[m * 16 * N_NEIGHBOR + bp];
      candv[m * 16 * N_NEIGHBOR + bp] = -1e30f;
    }
  }
  __syncthreads();

  // agg[m, :] = sum_k (val_k / 10) * kh[idx_k, :]  -> reuse stile
  {
    int m = tid >> 4, c0 = (tid & 15) * 16;
    for (int j = 0; j < 16; ++j) {
      float s = 0.f;
      for (int k = 0; k < N_NEIGHBOR; ++k)
        s += (topv[m * N_NEIGHBOR + k] * (1.f / N_NEIGHBOR)) *
             khf[(size_t)topi[m * N_NEIGHBOR + k] * HID + c0 + j];
      stile[m * HID + c0 + j] = s;
    }
  }
  __syncthreads();

  if (tid < 16) {
    int m = tid;
    float y = fcb[0];
    for (int c = 0; c < HID; ++c)
      y += mbf[(size_t)(m0 + m) * HID + c] * fcw[c] +
           stile[m * HID + c] * fcw[HID + c];
    out[m0 + m] = y;
  }
}

// ---------- host ----------
extern "C" void kernel_launch(void* const* d_in, const int* in_sizes, int n_in,
                              void* d_out, int out_size, void* d_ws, size_t ws_size,
                              hipStream_t stream) {
  const float* inp    = (const float*)d_in[0];
  const float* th     = (const float*)d_in[1];
  const float* thd    = (const float*)d_in[2];
  const float* W_ih0  = (const float*)d_in[3];
  const float* W_hh0  = (const float*)d_in[4];
  const float* b_ih0  = (const float*)d_in[5];
  const float* b_hh0  = (const float*)d_in[6];
  const float* W_ih1  = (const float*)d_in[7];
  const float* W_hh1  = (const float*)d_in[8];
  const float* b_ih1  = (const float*)d_in[9];
  const float* b_hh1  = (const float*)d_in[10];
  const float* lin0_W = (const float*)d_in[11];
  const float* lin0_b = (const float*)d_in[12];
  const float* lin1_W = (const float*)d_in[13];
  const float* lin1_b = (const float*)d_in[14];
  const float* lin2_W = (const float*)d_in[15];
  const float* lin2_b = (const float*)d_in[16];
  const float* proj1_W = (const float*)d_in[17];
  const float* proj2_W = (const float*)d_in[18];
  const float* fc_W   = (const float*)d_in[19];
  const float* fc_b   = (const float*)d_in[20];
  float* out = (float*)d_out;

  char* ws = (char*)d_ws;
  size_t off = 0;
  auto alloc = [&](size_t bytes) -> void* {
    void* p = ws + off;
    off = (off + bytes + 255) & ~(size_t)255;
    return p;
  };

  _Float16* hseq   = (_Float16*)alloc((size_t)N_SEQ * T_STEPS * HID * 2);
  _Float16* whh0_h = (_Float16*)alloc((size_t)G3 * HID * 2);
  _Float16* whh1_h = (_Float16*)alloc((size_t)G3 * HID * 2);
  _Float16* wih1_h = (_Float16*)alloc((size_t)G3 * HID * 2);
  _Float16* l0w_h  = (_Float16*)alloc((size_t)512 * HID * 2);
  _Float16* l1w_h  = (_Float16*)alloc((size_t)512 * 512 * 2);
  _Float16* l2w_h  = (_Float16*)alloc((size_t)HID * 512 * 2);
  _Float16* p1w_h  = (_Float16*)alloc((size_t)HID * HID * 2);
  _Float16* p2w_h  = (_Float16*)alloc((size_t)HID * HID * 2);
  _Float16* out16  = (_Float16*)alloc((size_t)N_SEQ * HID * 2);
  _Float16* mb1_h  = (_Float16*)alloc((size_t)N_SEQ * 512 * 2);
  _Float16* mb2_h  = (_Float16*)alloc((size_t)N_SEQ * 512 * 2);
  float*    mb_f   = (float*)alloc((size_t)N_SEQ * HID * 4);
  _Float16* mb_h   = (_Float16*)alloc((size_t)N_SEQ * HID * 2);
  float*    q_f    = (float*)alloc((size_t)N_SEQ * HID * 4);
  _Float16* q_h    = (_Float16*)alloc((size_t)N_SEQ * HID * 2);
  float*    qn     = (float*)alloc((size_t)N_SEQ * 4);
  int*      didx   = (int*)alloc((size_t)B_DAYS * K_DAY * 4);
  _Float16* samp_h = (_Float16*)alloc((size_t)N_SAMPLE * HID * 2);
  float*    kh_f   = (float*)alloc((size_t)N_SAMPLE * HID * 4);
  _Float16* kh_h   = (_Float16*)alloc((size_t)N_SAMPLE * HID * 2);
  float*    khn    = (float*)alloc((size_t)N_SAMPLE * 4);

  // weight conversions
  auto conv = [&](const float* s, _Float16* d, int n) {
    k_f32_to_f16<<<(n + 255) / 256, 256, 0, stream>>>(s, d, n);
  };
  conv(W_hh0, whh0_h, G3 * HID);
  conv(W_hh1, whh1_h, G3 * HID);
  conv(W_ih1, wih1_h, G3 * HID);
  conv(lin0_W, l0w_h, 512 * HID);
  conv(lin1_W, l1w_h, 512 * 512);
  conv(lin2_W, l2w_h, HID * 512);
  conv(proj1_W, p1w_h, HID * HID);
  conv(proj2_W, p2w_h, HID * HID);

  // GRU layers
  k_gru0<<<N_SEQ / 16, 256, 8192 + 16384 + 49152, stream>>>(
      inp, whh0_h, W_ih0, b_ih0, b_hh0, hseq);
  k_gru1<<<N_SEQ / 16, 256, 8192 + 16384 + 49152 + 49152, stream>>>(
      hseq, wih1_h, whh1_h, b_ih1, b_hh1, out16);

  // MLP (leaky)
  k_gemm16<1><<<dim3(N_SEQ / 16, 4), 256, 0, stream>>>(
      out16, l0w_h, HID, lin0_b, nullptr, mb1_h, 512);
  k_gemm16<1><<<dim3(N_SEQ / 16, 4), 256, 0, stream>>>(
      mb1_h, l1w_h, 512, lin1_b, nullptr, mb2_h, 512);
  k_gemm16<1><<<dim3(N_SEQ / 16, 2), 256, 0, stream>>>(
      mb2_h, l2w_h, 512, lin2_b, mb_f, mb_h, HID);

  // day-level top-k and gather
  k_day_topk<<<B_DAYS, 256, 0, stream>>>(mb_f, thd, didx);
  k_gather<<<(N_SAMPLE * HID) / 256, 256, 0, stream>>>(th, didx, samp_h);

  // projections
  k_gemm16<0><<<dim3(N_SEQ / 16, 2), 256, 0, stream>>>(
      mb_h, p1w_h, HID, nullptr, q_f, q_h, HID);
  k_gemm16<0><<<dim3(N_SAMPLE / 16, 2), 256, 0, stream>>>(
      samp_h, p2w_h, HID, nullptr, kh_f, kh_h, HID);

  // norms
  k_rownorm<<<(N_SEQ + 255) / 256, 256, 0, stream>>>(q_f, qn, HID, N_SEQ);
  k_rownorm<<<(N_SAMPLE + 255) / 256, 256, 0, stream>>>(kh_f, khn, HID, N_SAMPLE);

  // streaming cosine top-k + aggregation + final FC
  k_topk_agg<<<N_SEQ / 16, 256, 0, stream>>>(
      q_h, kh_h, kh_f, qn, khn, mb_f, fc_W, fc_b, out);
}
